// MultiViewPredictor_30726196035718
// MI455X (gfx1250) — matrix operand order repaired and checked by reading, not compiled, not run on hardware
//
#include <hip/hip_runtime.h>
#include <hip/hip_bf16.h>
#include <math.h>

typedef __attribute__((ext_vector_type(16))) _Float16 v16h;
typedef __attribute__((ext_vector_type(8)))  float    v8f;

#define BB 4
#define NL 20          // N labeled stations
#define TT 28
#define VV 3
#define NN 2048        // NNODES
#define F_MEO 8
#define F_ST_UN 12
#define F_ST_LB 13
#define F_NODE 16
#define HH 32
#define OUT_STEPS 7
#define PREV_SLOT 12
#define TILES 2        // ceil(NL/16)

__device__ __forceinline__ float sigf(float x) { return 1.0f / (1.0f + __expf(-x)); }
__device__ __forceinline__ float lrelu(float x) { return x >= 0.f ? x : 0.2f * x; }
__device__ __forceinline__ float eluf(float x) { return x > 0.f ? x : (__expf(x) - 1.f); }

// ---------------- K1a: Wh = node_feat @ gat_W (f32 row-major + f16 transposed) ----------
// Wh16T layout: [b][h][node] so a WMMA B-fragment (16 consecutive K for fixed n)
// is one contiguous 32-byte load.
__global__ void k_wh(const float* __restrict__ nf, const float* __restrict__ gatW,
                     float* __restrict__ Wh32, _Float16* __restrict__ Wh16T) {
    int idx = blockIdx.x * blockDim.x + threadIdx.x;
    if (idx >= BB * NN * HH) return;
    int h = idx % HH;
    int bn = idx / HH;
    int b = bn / NN, node = bn % NN;
    const float* x = nf + (size_t)bn * F_NODE;
    float acc = 0.f;
#pragma unroll
    for (int f = 0; f < F_NODE; ++f) acc += x[f] * gatW[f * HH + h];
    Wh32[idx] = acc;
    Wh16T[((size_t)b * HH + h) * NN + node] = (_Float16)acc;
}

// ---------------- K1b: f1/f2 = Wh @ a1 / a2 ----------------
__global__ void k_f12(const float* __restrict__ Wh32, const float* __restrict__ a1,
                      const float* __restrict__ a2, float* __restrict__ f1,
                      float* __restrict__ f2) {
    int idx = blockIdx.x * blockDim.x + threadIdx.x;
    if (idx >= BB * NN) return;
    const float* r = Wh32 + (size_t)idx * HH;
    float s1 = 0.f, s2 = 0.f;
#pragma unroll
    for (int h = 0; h < HH; ++h) { s1 += r[h] * a1[h]; s2 += r[h] * a2[h]; }
    f1[idx] = s1;
    f2[idx] = s2;
}

// ---------------- K1c: per-batch max over f2 (single-pass softmax shift) ----------------
__global__ void k_f2max(const float* __restrict__ f2, float* __restrict__ f2max) {
    int b = blockIdx.x, tid = threadIdx.x;
    __shared__ float red[256];
    float m = -1e30f;
    for (int j = tid; j < NN; j += 256) m = fmaxf(m, f2[b * NN + j]);
    red[tid] = m;
    __syncthreads();
    for (int s = 128; s > 0; s >>= 1) {
        if (tid < s) red[tid] = fmaxf(red[tid], red[tid + s]);
        __syncthreads();
    }
    if (tid == 0) f2max[b] = red[0];
}

// ---------------- K2: fused GAT attention + aggregation + multi-view fusion (WMMA) -------
// One wave per (b, row-tile of 16 gathered stations). K-chunk outer loop:
// exp(leaky(f1_i+f2_j)-shift_i) and the Wh B-fragments are view-invariant and
// computed once; per view only the adjacency mask is applied (branch-free
// cndmask) before the two v_wmma_f32_16x16x32_f16 per view. Softmax 1/rowsum,
// ELU and the view-fusion coefficient are folded into the C fragments.
__global__ __launch_bounds__(32) void k_gat(
    const float* __restrict__ adj, const int* __restrict__ gid,
    const _Float16* __restrict__ Wh16T, const float* __restrict__ f1,
    const float* __restrict__ f2, const float* __restrict__ f2max,
    const float* __restrict__ vlog, float* __restrict__ fusedg) {
    const int blk = blockIdx.x;
    const int b = blk / TILES, tile = blk % TILES;
    const int lane = threadIdx.x;
    const int mrow = lane & 15;   // A-matrix row / C-matrix column index
    const int hi = lane >> 4;     // K-half selector

    __shared__ int   rows_s[16];
    __shared__ float f1_s[16], shift_s[16];
    __shared__ float partial[32];
    __shared__ float rowsum[16];

    // multi-view fusion coefficients: alpha*softmax(logits) + (1-alpha)/V
    float l0 = vlog[0], l1 = vlog[1], l2 = vlog[2];
    float lmx = fmaxf(l0, fmaxf(l1, l2));
    float e0 = __expf(l0 - lmx), e1 = __expf(l1 - lmx), e2 = __expf(l2 - lmx);
    float einv = 1.f / (e0 + e1 + e2);
    float coeff[VV] = { 0.6f * e0 * einv + 0.4f / 3.f,
                        0.6f * e1 * einv + 0.4f / 3.f,
                        0.6f * e2 * einv + 0.4f / 3.f };

    if (lane < 16) {
        int mg = tile * 16 + lane;
        int row = (mg < NL) ? gid[b * NL + mg] : gid[b * NL];  // pad rows: dup row 0
        rows_s[lane] = row;
        float f1v = f1[b * NN + row];
        f1_s[lane] = f1v;
        shift_s[lane] = lrelu(f1v + f2max[b]);  // upper bound of masked e_ij: safe shift
    }
    __syncthreads();

    const int row = rows_s[mrow];
    const float f1v = f1_s[mrow];
    const float sh = shift_s[mrow];
    const float* f2b = f2 + b * NN;
    const _Float16* WhT0 = Wh16T + ((size_t)b * HH + mrow) * NN;
    const _Float16* WhT1 = Wh16T + ((size_t)b * HH + mrow + 16) * NN;
    const size_t adjBase = (size_t)b * VV * NN * NN + (size_t)row * NN;
    const float* adjRow[VV];
#pragma unroll
    for (int v = 0; v < VV; ++v) adjRow[v] = adj + adjBase + (size_t)v * NN * NN;

    v8f cacc0[VV], cacc1[VV];
    float psum[VV];
#pragma unroll
    for (int v = 0; v < VV; ++v) { cacc0[v] = (v8f){}; cacc1[v] = (v8f){}; psum[v] = 0.f; }

    for (int kb = 0; kb < NN; kb += 32) {
        // ---- view-invariant: f2 chunk + exp(leaky(..)-shift), B fragments ----
        float fv[16], pexp[16];
        const float* fj0 = f2b + kb + 16 * hi;       // A layout: K=e+16*hi, contiguous
        const float* fj1 = f2b + kb + 16 * hi + 8;
        *(float4*)&fv[0]  = *(const float4*)(fj0);
        *(float4*)&fv[4]  = *(const float4*)(fj0 + 4);
        *(float4*)&fv[8]  = *(const float4*)(fj1);
        *(float4*)&fv[12] = *(const float4*)(fj1 + 4);
#pragma unroll
        for (int e = 0; e < 16; ++e) pexp[e] = __expf(lrelu(f1v + fv[e]) - sh);

        v16h bf0 = *(const v16h*)(WhT0 + kb + 16 * hi);  // B layout: K=e+16*hi, contiguous
        v16h bf1 = *(const v16h*)(WhT1 + kb + 16 * hi);

        // ---- per view: adjacency mask (branch-free) + 2 WMMAs ----
#pragma unroll
        for (int v = 0; v < VV; ++v) {
            __builtin_prefetch((const char*)(adjRow[v] + kb) + 4096, 0, 1);
            float av[16];
            const float* aj0 = adjRow[v] + kb + 16 * hi;
            const float* aj1 = adjRow[v] + kb + 16 * hi + 8;
            *(float4*)&av[0]  = *(const float4*)(aj0);
            *(float4*)&av[4]  = *(const float4*)(aj0 + 4);
            *(float4*)&av[8]  = *(const float4*)(aj1);
            *(float4*)&av[12] = *(const float4*)(aj1 + 4);
            v16h a;
            float ps = 0.f;
#pragma unroll
            for (int e = 0; e < 16; ++e) {
                float msk = (av[e] > 0.f) ? 1.f : 0.f;   // v_cndmask, no EXEC games
                float p = msk * pexp[e];
                ps += p;
                a[e] = (_Float16)p;
            }
            psum[v] += ps;
            cacc0[v] = __builtin_amdgcn_wmma_f32_16x16x32_f16(false, a, false, bf0,
                                                              (short)0, cacc0[v], false, false);
            cacc1[v] = __builtin_amdgcn_wmma_f32_16x16x32_f16(false, a, false, bf1,
                                                              (short)0, cacc1[v], false, false);
        }
    }

    // ---- per view: softmax denominator, ELU, view-fusion coefficient ----
    v8f acc0 = {}, acc1 = {};
#pragma unroll
    for (int v = 0; v < VV; ++v) {
        partial[lane] = psum[v];
        __syncthreads();
        if (lane < 16) rowsum[lane] = partial[lane] + partial[lane + 16];
        __syncthreads();
        float cv = coeff[v];
#pragma unroll
        for (int r = 0; r < 8; ++r) {
            int mr = r + 8 * hi;  // C-fragment row for this lane/vgpr
            float rs = rowsum[mr];
            rs = (rs > 0.f) ? rs : 1.f;
            acc0[r] = acc0[r] + cv * eluf(cacc0[v][r] / rs);
            acc1[r] = acc1[r] + cv * eluf(cacc1[v][r] / rs);
        }
        __syncthreads();
    }

    // store fused gathered rows: C layout row=r+8*hi, col=lane&15
#pragma unroll
    for (int r = 0; r < 8; ++r) {
        int mr = r + 8 * hi;
        int mg = tile * 16 + mr;
        if (mg < NL) {
            float* dst = fusedg + ((size_t)(b * NL + mg)) * HH;
            dst[mrow] = acc0[r];
            dst[mrow + 16] = acc1[r];
        }
    }
}

// ---------------- K3: g = relu(fused @ mv_W + mv_b) at gathered rows ----------------
__global__ void k_gmv(const float* __restrict__ fusedg, const float* __restrict__ mvW,
                      const float* __restrict__ mvb, float* __restrict__ g) {
    int idx = blockIdx.x * blockDim.x + threadIdx.x;
    if (idx >= BB * NL * HH) return;
    int h = idx % HH;
    int rw = idx / HH;
    const float* fr = fusedg + (size_t)rw * HH;
    float acc = mvb[h];
#pragma unroll
    for (int k = 0; k < HH; ++k) acc += fr[k] * mvW[k * HH + h];
    g[idx] = fmaxf(acc, 0.f);
}

// ---------------- K4: unlabeled branch (LSTM + lin1 + lin2) ----------------
__global__ __launch_bounds__(64) void k_ul(
    const float* __restrict__ meo_ul, const float* __restrict__ feat_ul,
    const float* __restrict__ Wih, const float* __restrict__ Whh,
    const float* __restrict__ bih, const float* __restrict__ bhh,
    const float* __restrict__ lin1W, const float* __restrict__ lin1b,
    const float* __restrict__ lin2W, const float* __restrict__ lin2b,
    float* __restrict__ unlabel) {
    int b = blockIdx.x, tid = threadIdx.x;
    __shared__ float hbuf[HH];
    __shared__ float cat[2 * HH];
    float c = 0.f, h = 0.f;
    if (tid < HH) hbuf[tid] = 0.f;
    __syncthreads();
    for (int t = 0; t < TT; ++t) {
        if (tid < HH) {
            const float* x = meo_ul + ((size_t)b * TT + t) * F_MEO;
            float gi = bih[tid] + bhh[tid];
            float gf = bih[HH + tid] + bhh[HH + tid];
            float gg = bih[2 * HH + tid] + bhh[2 * HH + tid];
            float go = bih[3 * HH + tid] + bhh[3 * HH + tid];
            for (int f = 0; f < F_MEO; ++f) {
                float xv = x[f];
                gi += xv * Wih[(0 * HH + tid) * F_MEO + f];
                gf += xv * Wih[(1 * HH + tid) * F_MEO + f];
                gg += xv * Wih[(2 * HH + tid) * F_MEO + f];
                go += xv * Wih[(3 * HH + tid) * F_MEO + f];
            }
            for (int k = 0; k < HH; ++k) {
                float hv = hbuf[k];
                gi += hv * Whh[(0 * HH + tid) * HH + k];
                gf += hv * Whh[(1 * HH + tid) * HH + k];
                gg += hv * Whh[(2 * HH + tid) * HH + k];
                go += hv * Whh[(3 * HH + tid) * HH + k];
            }
            c = sigf(gf) * c + sigf(gi) * tanhf(gg);
            h = sigf(go) * tanhf(c);
        }
        __syncthreads();
        if (tid < HH) hbuf[tid] = h;
        __syncthreads();
    }
    if (tid < HH) {
        cat[tid] = h;
    } else {
        int j = tid - HH;
        const float* fu = feat_ul + b * F_ST_UN;
        float acc = lin1b[j];
        for (int f = 0; f < F_ST_UN; ++f) acc += fu[f] * lin1W[f * HH + j];
        cat[tid] = fmaxf(acc, 0.f);
    }
    __syncthreads();
    float acc = lin2b[tid];
    for (int k = 0; k < 2 * HH; ++k) acc += cat[k] * lin2W[k * (2 * HH) + tid];
    unlabel[b * (2 * HH) + tid] = fmaxf(acc, 0.f);
}

// ---------------- K5: labeled branch (LSTM + lin1 + concat g + lin2) ----------------
__global__ __launch_bounds__(96) void k_lb(
    const float* __restrict__ ovi, const float* __restrict__ meo,
    const float* __restrict__ feat_lb, const float* __restrict__ g,
    const float* __restrict__ Wih, const float* __restrict__ Whh,
    const float* __restrict__ bih, const float* __restrict__ bhh,
    const float* __restrict__ lin1W, const float* __restrict__ lin1b,
    const float* __restrict__ lin2W, const float* __restrict__ lin2b,
    float* __restrict__ label) {
    const int F_IN = 1 + F_MEO;  // 9
    int bn = blockIdx.x, tid = threadIdx.x;
    __shared__ float hbuf[HH];
    __shared__ float cat[3 * HH];
    float c = 0.f, h = 0.f;
    if (tid < HH) hbuf[tid] = 0.f;
    __syncthreads();
    for (int t = 0; t < TT; ++t) {
        if (tid < HH) {
            float x0 = ovi[(size_t)bn * TT + t];
            const float* mrow = meo + ((size_t)bn * TT + t) * F_MEO;
            float gi = bih[tid] + bhh[tid];
            float gf = bih[HH + tid] + bhh[HH + tid];
            float gg = bih[2 * HH + tid] + bhh[2 * HH + tid];
            float go = bih[3 * HH + tid] + bhh[3 * HH + tid];
            gi += x0 * Wih[(0 * HH + tid) * F_IN];
            gf += x0 * Wih[(1 * HH + tid) * F_IN];
            gg += x0 * Wih[(2 * HH + tid) * F_IN];
            go += x0 * Wih[(3 * HH + tid) * F_IN];
            for (int f = 0; f < F_MEO; ++f) {
                float xv = mrow[f];
                gi += xv * Wih[(0 * HH + tid) * F_IN + 1 + f];
                gf += xv * Wih[(1 * HH + tid) * F_IN + 1 + f];
                gg += xv * Wih[(2 * HH + tid) * F_IN + 1 + f];
                go += xv * Wih[(3 * HH + tid) * F_IN + 1 + f];
            }
            for (int k = 0; k < HH; ++k) {
                float hv = hbuf[k];
                gi += hv * Whh[(0 * HH + tid) * HH + k];
                gf += hv * Whh[(1 * HH + tid) * HH + k];
                gg += hv * Whh[(2 * HH + tid) * HH + k];
                go += hv * Whh[(3 * HH + tid) * HH + k];
            }
            c = sigf(gf) * c + sigf(gi) * tanhf(gg);
            h = sigf(go) * tanhf(c);
        }
        __syncthreads();
        if (tid < HH) hbuf[tid] = h;
        __syncthreads();
    }
    if (tid < HH) {
        cat[tid] = h;
    } else if (tid < 2 * HH) {
        int j = tid - HH;
        const float* fl = feat_lb + bn * F_ST_LB;
        float acc = lin1b[j];
        for (int f = 0; f < F_ST_LB; ++f) acc += fl[f] * lin1W[f * HH + j];
        cat[tid] = fmaxf(acc, 0.f);
    } else {
        cat[tid] = g[bn * HH + (tid - 2 * HH)];
    }
    __syncthreads();
    if (tid < 2 * HH) {
        float acc = lin2b[tid];
        for (int k = 0; k < 3 * HH; ++k) acc += cat[k] * lin2W[k * (2 * HH) + tid];
        label[bn * (2 * HH) + tid] = fmaxf(acc, 0.f);
    }
}

// ---------------- K6: inverse-distance-weighted attention pooling ----------------
__global__ __launch_bounds__(128) void k_att(
    const float* __restrict__ unlabel, const float* __restrict__ label,
    const float* __restrict__ Wq, const float* __restrict__ Wk,
    const float* __restrict__ dis, float* __restrict__ sp) {
    int b = blockIdx.x, tid = threadIdx.x;
    __shared__ float q[16];
    __shared__ float score[NL];
    if (tid < 16) {
        float acc = 0.f;
        for (int cc = 0; cc < 2 * HH; ++cc) acc += unlabel[b * 2 * HH + cc] * Wq[cc * 16 + tid];
        q[tid] = acc;
    }
    __syncthreads();
    if (tid < NL) {
        const float* ld = label + (size_t)(b * NL + tid) * 2 * HH;
        float lg = 0.f;
        for (int d = 0; d < 16; ++d) {
            float kd = 0.f;
            for (int cc = 0; cc < 2 * HH; ++cc) kd += ld[cc] * Wk[cc * 16 + d];
            lg += kd * q[d];
        }
        score[tid] = lg * 0.25f;
    }
    __syncthreads();
    if (tid == 0) {
        float m = -1e30f;
        for (int n = 0; n < NL; ++n) m = fmaxf(m, score[n]);
        float se = 0.f;
        for (int n = 0; n < NL; ++n) { float e = __expf(score[n] - m); score[n] = e; se += e; }
        float ti = 0.f;
        for (int n = 0; n < NL; ++n) {
            float iw = (score[n] / se) / (dis[b * NL + n] + 1e-6f);
            score[n] = iw;
            ti += iw;
        }
        float rinv = 1.f / ti;
        for (int n = 0; n < NL; ++n) score[n] *= rinv;
    }
    __syncthreads();
    if (tid < 2 * HH) {
        float acc = 0.f;
        for (int n = 0; n < NL; ++n) acc += score[n] * label[(size_t)(b * NL + n) * 2 * HH + tid];
        sp[b * 2 * HH + tid] = fmaxf(acc, 0.f);
    }
}

// ---------------- K7: 12-step GRU (zero hidden each step) + output head ----------------
__global__ __launch_bounds__(96) void k_gru(
    const float* __restrict__ unlabel, const float* __restrict__ sp,
    const float* __restrict__ h_t, const float* __restrict__ Wih,
    const float* __restrict__ bih, const float* __restrict__ bhh,
    const float* __restrict__ lintW, const float* __restrict__ lintb,
    const float* __restrict__ outW, const float* __restrict__ outb,
    float* __restrict__ out) {
    int b = blockIdx.x, tid = threadIdx.x;
    __shared__ float hbuf[HH], temp[HH], gbuf[3 * HH];
    const float* Wr = Wih + (size_t)tid * 5 * HH;  // X = [unlabel(64), temp(32), sp(64)]
    float cst = bih[tid];
    for (int cc = 0; cc < 2 * HH; ++cc) cst += unlabel[b * 2 * HH + cc] * Wr[cc];
    for (int cc = 0; cc < 2 * HH; ++cc) cst += sp[b * 2 * HH + cc] * Wr[3 * HH + cc];
    if (tid < HH) hbuf[tid] = h_t[b * HH + tid];
    __syncthreads();
    for (int s = 0; s < PREV_SLOT; ++s) {
        if (tid < HH) {
            float acc = lintb[tid];
            for (int k = 0; k < HH; ++k) acc += hbuf[k] * lintW[k * HH + tid];
            temp[tid] = fmaxf(acc, 0.f);
        }
        __syncthreads();
        float gv = cst;
        for (int k = 0; k < HH; ++k) gv += temp[k] * Wr[2 * HH + k];
        gbuf[tid] = gv;
        __syncthreads();
        if (tid < HH) {
            float r = sigf(gbuf[tid] + bhh[tid]);
            float z = sigf(gbuf[HH + tid] + bhh[HH + tid]);
            float nn = tanhf(gbuf[2 * HH + tid] + r * bhh[2 * HH + tid]);
            hbuf[tid] = (1.f - z) * nn;  // h_prev = 0, so z*h drops
        }
        __syncthreads();
    }
    if (tid < OUT_STEPS) {
        float acc = outb[tid];
        for (int k = 0; k < HH; ++k) acc += hbuf[k] * outW[k * OUT_STEPS + tid];
        out[b * OUT_STEPS + tid] = acc;
    }
}

extern "C" void kernel_launch(void* const* d_in, const int* in_sizes, int n_in,
                              void* d_out, int out_size, void* d_ws, size_t ws_size,
                              hipStream_t stream) {
    // tensor inputs (setup_inputs dict order)
    const float* meo_unlabel    = (const float*)d_in[0];
    const float* feature_unlabel= (const float*)d_in[1];
    const float* ovi_label      = (const float*)d_in[2];
    const float* meo_label      = (const float*)d_in[3];
    const float* feature_label  = (const float*)d_in[4];
    const float* dis_label      = (const float*)d_in[5];
    const float* h_t            = (const float*)d_in[6];
    const float* node_feat      = (const float*)d_in[7];
    const float* adj            = (const float*)d_in[8];
    const int*   gid_idx        = (const int*)  d_in[9];
    // params: JAX pytree flattening sorts dict keys (ASCII order)
    const float* att_Wk   = (const float*)d_in[10];
    const float* att_Wq   = (const float*)d_in[11];
    const float* gat_W    = (const float*)d_in[12];
    const float* gat_a1   = (const float*)d_in[13];
    const float* gat_a2   = (const float*)d_in[14];
    const float* gru_Wih  = (const float*)d_in[15];
    const float* gru_bhh  = (const float*)d_in[16];
    const float* gru_bih  = (const float*)d_in[17];
    const float* lb_lin1_W  = (const float*)d_in[18];
    const float* lb_lin1_b  = (const float*)d_in[19];
    const float* lb_lin2_W  = (const float*)d_in[20];
    const float* lb_lin2_b  = (const float*)d_in[21];
    const float* lb_lstm_Whh= (const float*)d_in[22];
    const float* lb_lstm_Wih= (const float*)d_in[23];
    const float* lb_lstm_bhh= (const float*)d_in[24];
    const float* lb_lstm_bih= (const float*)d_in[25];
    const float* lint_W   = (const float*)d_in[26];
    const float* lint_b   = (const float*)d_in[27];
    const float* mv_W     = (const float*)d_in[28];
    const float* mv_b     = (const float*)d_in[29];
    const float* mv_vlog  = (const float*)d_in[30];
    const float* out_W    = (const float*)d_in[31];
    const float* out_b    = (const float*)d_in[32];
    const float* ul_lin1_W  = (const float*)d_in[33];
    const float* ul_lin1_b  = (const float*)d_in[34];
    const float* ul_lin2_W  = (const float*)d_in[35];
    const float* ul_lin2_b  = (const float*)d_in[36];
    const float* ul_lstm_Whh= (const float*)d_in[37];
    const float* ul_lstm_Wih= (const float*)d_in[38];
    const float* ul_lstm_bhh= (const float*)d_in[39];
    const float* ul_lstm_bih= (const float*)d_in[40];
    float* out = (float*)d_out;

    // workspace carve-up
    char* ws = (char*)d_ws;
    float*    Wh32   = (float*)ws;     ws += (size_t)BB * NN * HH * sizeof(float);
    _Float16* Wh16T  = (_Float16*)ws;  ws += (size_t)BB * NN * HH * sizeof(_Float16);
    float*    f1     = (float*)ws;     ws += (size_t)BB * NN * sizeof(float);
    float*    f2     = (float*)ws;     ws += (size_t)BB * NN * sizeof(float);
    float*    f2max  = (float*)ws;     ws += 16 * sizeof(float);
    float*    fusedg = (float*)ws;     ws += (size_t)BB * NL * HH * sizeof(float);
    float*    g      = (float*)ws;     ws += (size_t)BB * NL * HH * sizeof(float);
    float*    unlab  = (float*)ws;     ws += (size_t)BB * 2 * HH * sizeof(float);
    float*    label  = (float*)ws;     ws += (size_t)BB * NL * 2 * HH * sizeof(float);
    float*    sp     = (float*)ws;     ws += (size_t)BB * 2 * HH * sizeof(float);

    (void)in_sizes; (void)n_in; (void)out_size; (void)ws_size;

    k_wh<<<(BB * NN * HH + 255) / 256, 256, 0, stream>>>(node_feat, gat_W, Wh32, Wh16T);
    k_f12<<<(BB * NN + 255) / 256, 256, 0, stream>>>(Wh32, gat_a1, gat_a2, f1, f2);
    k_f2max<<<BB, 256, 0, stream>>>(f2, f2max);
    k_gat<<<BB * TILES, 32, 0, stream>>>(adj, gid_idx, Wh16T, f1, f2, f2max, mv_vlog, fusedg);
    k_gmv<<<(BB * NL * HH + 255) / 256, 256, 0, stream>>>(fusedg, mv_W, mv_b, g);
    k_ul<<<BB, 64, 0, stream>>>(meo_unlabel, feature_unlabel, ul_lstm_Wih, ul_lstm_Whh,
                                ul_lstm_bih, ul_lstm_bhh, ul_lin1_W, ul_lin1_b,
                                ul_lin2_W, ul_lin2_b, unlab);
    k_lb<<<BB * NL, 96, 0, stream>>>(ovi_label, meo_label, feature_label, g,
                                     lb_lstm_Wih, lb_lstm_Whh, lb_lstm_bih, lb_lstm_bhh,
                                     lb_lin1_W, lb_lin1_b, lb_lin2_W, lb_lin2_b, label);
    k_att<<<BB, 128, 0, stream>>>(unlab, label, att_Wq, att_Wk, dis_label, sp);
    k_gru<<<BB, 96, 0, stream>>>(unlab, sp, h_t, gru_Wih, gru_bih, gru_bhh,
                                 lint_W, lint_b, out_W, out_b, out);
}
